// GQAAttention_23570780520686
// MI455X (gfx1250) — matrix operand order repaired and checked by reading, not compiled, not run on hardware
//
#include <hip/hip_runtime.h>

// ---------------------------------------------------------------- types
typedef unsigned short u16;
typedef __attribute__((ext_vector_type(16))) __bf16 v16bf;
typedef __attribute__((ext_vector_type(8)))  float  v8f;

union FragAB {
  v16bf v;
  uint4 q[2];
};

// ---------------------------------------------------------------- consts
constexpr int BB   = 2;
constexpr int S    = 2048;
constexpr int HID  = 2048;
constexpr int NH   = 16;
constexpr int NKV  = 4;
constexpr int HD   = 128;
constexpr int ROWS = BB * S;              // 4096
constexpr int NQKV = HID + 2 * NKV * HD;  // 3072 (Q|K|V concat cols)

// ---------------------------------------------------------------- helpers
__device__ __forceinline__ u16 f2bf(float f) {
  unsigned int u = __float_as_uint(f);
  u += 0x7FFFu + ((u >> 16) & 1u);           // round-to-nearest-even
  return (u16)(u >> 16);
}
__device__ __forceinline__ float bf2f(u16 h) {
  return __uint_as_float(((unsigned int)h) << 16);
}

__device__ __forceinline__ v8f wmma_bf16(v16bf a, v16bf b, v8f c) {
  // (neg_a, A, neg_b, B, c_mod, C, reuse_a, reuse_b)
  return __builtin_amdgcn_wmma_f32_16x16x32_bf16(false, a, false, b, (short)0, c,
                                                 false, false);
}

// A-fragment (16x32 bf16): lane row M = lane%16; elem e -> col
//   k = kbase + 16*(lane/16) + e   => one 32B contiguous run per lane
__device__ __forceinline__ v16bf load_a_frag(const u16* row, int kbase, int half) {
  FragAB f;
  f.q[0] = *(const uint4*)(row + kbase + half * 16);
  f.q[1] = *(const uint4*)(row + kbase + half * 16 + 8);
  return f.v;
}
// B-fragment (32x16 bf16): lane col N = lane%16; elem e -> k = kbase + 16*(lane/16) + e
//   => one 32B contiguous run per lane
__device__ __forceinline__ v16bf load_b_frag(const u16* colrow, int kbase, int half) {
  FragAB f;
  f.q[0] = *(const uint4*)(colrow + kbase + half * 16);
  f.q[1] = *(const uint4*)(colrow + kbase + half * 16 + 8);
  return f.v;
}

// exchange value with lane^16 (wave32) : ds_swizzle SWAPX16 (and=0x1f, xor=0x10)
__device__ __forceinline__ float xor16f(float x) {
  return __int_as_float(__builtin_amdgcn_ds_swizzle(__float_as_int(x), 0x401F));
}

// CDNA5 async copy: 16 bytes global -> LDS, tracked by ASYNCcnt.
// vdst = LDS byte address (low 32 bits of the generic shared pointer),
// vaddr = 64-bit global address, GV addressing mode.
__device__ __forceinline__ void async_copy16(const u16* g, u16* lds) {
  unsigned       laddr = (unsigned)(unsigned long long)(size_t)lds;
  unsigned long long ga = (unsigned long long)(size_t)g;
  asm volatile("global_load_async_to_lds_b128 %0, %1, off"
               :
               : "v"(laddr), "v"(ga)
               : "memory");
}
__device__ __forceinline__ void async_wait_le4() {
  asm volatile("s_wait_asynccnt 0x4" ::: "memory");
}

// ---------------------------------------------------------------- conversion kernels
__global__ void __launch_bounds__(256) k_f2bf(const float* __restrict__ src,
                                              u16* __restrict__ dst, size_t n) {
  for (size_t i = (size_t)blockIdx.x * blockDim.x + threadIdx.x; i < n;
       i += (size_t)gridDim.x * blockDim.x)
    dst[i] = f2bf(src[i]);
}

// src[K][N] f32  ->  dst[N][K] bf16
__global__ void __launch_bounds__(256) k_transpose(const float* __restrict__ src,
                                                   u16* __restrict__ dst, int K, int N) {
  size_t total = (size_t)K * N;
  for (size_t i = (size_t)blockIdx.x * blockDim.x + threadIdx.x; i < total;
       i += (size_t)gridDim.x * blockDim.x) {
    size_t k = i / (size_t)N, n = i % (size_t)N;
    dst[n * (size_t)K + k] = f2bf(src[i]);
  }
}

// ---------------------------------------------------------------- GEMM (bf16 x bf16 -> f32 acc)
// C[ROWS x NCOLS] = A[ROWS x HID] * B ; Bt holds B^T rows (= output columns).
// Per wave: 16M x 64N tile (4 accumulators share one A-fragment per k-step).
template <int NCOLS, bool OUT_F32>
__global__ void __launch_bounds__(256) k_gemm(const u16* __restrict__ A,
                                              const u16* __restrict__ Bt,
                                              void* __restrict__ OutV) {
  const int tid = threadIdx.x;
  const int wave = tid >> 5, lane = tid & 31;
  const int half = lane >> 4, l16 = lane & 15;
  constexpr int NGRP = NCOLS / 512;  // 8 waves * 64 cols
  const int mtile = blockIdx.x / NGRP;
  const int ngrp  = blockIdx.x % NGRP;
  const int m0 = mtile * 16, n0 = ngrp * 512 + wave * 64;

  const u16* arow = A + (size_t)(m0 + l16) * HID;
  const u16* brow[4];
#pragma unroll
  for (int j = 0; j < 4; ++j) brow[j] = Bt + (size_t)(n0 + j * 16 + l16) * HID;

  v8f acc[4] = {};
#pragma unroll 2
  for (int kk = 0; kk < HID; kk += 32) {
    const v16bf af = load_a_frag(arow, kk, half);
#pragma unroll
    for (int j = 0; j < 4; ++j)
      acc[j] = wmma_bf16(af, load_b_frag(brow[j], kk, half), acc[j]);
  }

#pragma unroll
  for (int j = 0; j < 4; ++j) {
#pragma unroll
    for (int r = 0; r < 8; ++r) {
      const size_t idx = (size_t)(m0 + r + 8 * half) * NCOLS + n0 + j * 16 + l16;
      if (OUT_F32)
        ((float*)OutV)[idx] = acc[j][r];
      else
        ((u16*)OutV)[idx] = f2bf(acc[j][r]);
    }
  }
}

// ---------------------------------------------------------------- RoPE + relayout
// QKVr [B,S,3072]bf16 -> Qb[B,NH,S,HD], Kb[B,NKV,S,HD] (with RoPE), Vt[B,NKV,HD,S]
__global__ void __launch_bounds__(256) k_rope(const u16* __restrict__ QKVr,
                                              const float* __restrict__ Cs,
                                              const float* __restrict__ Sn,
                                              u16* __restrict__ Qb,
                                              u16* __restrict__ Kb,
                                              u16* __restrict__ Vt) {
  const int s = blockIdx.x & (S - 1);
  const int b = blockIdx.x >> 11;
  const u16* row = QKVr + ((size_t)(b * S + s)) * NQKV;
  const float* cs = Cs + (size_t)s * HD;
  const float* sn = Sn + (size_t)s * HD;
  for (int i = threadIdx.x; i < NQKV; i += 256) {
    const int h = i >> 7, d = i & (HD - 1);
    if (i < HID + NKV * HD) {  // Q or K : rotary
      const float x  = bf2f(row[i]);
      const int   pj = (d < 64) ? (i + 64) : (i - 64);
      const float xp = bf2f(row[pj]);
      const float r  = (d < 64) ? (x * cs[d] - xp * sn[d]) : (x * cs[d] + xp * sn[d]);
      const u16   o  = f2bf(r);
      if (i < HID)
        Qb[((size_t)(b * NH + h) * S + s) * HD + d] = o;
      else
        Kb[((size_t)(b * NKV + (h - NH)) * S + s) * HD + d] = o;
    } else {  // V : copy transposed
      const int vh = h - (NH + NKV);
      Vt[((size_t)(b * NKV + vh) * HD + d) * S + s] = row[i];
    }
  }
}

// ---------------------------------------------------------------- flash attention
// Block = one (batch, head, 128-query group); 8 waves, one 16-query tile each.
// K/V 32-key tiles double-buffered in LDS via async global->LDS DMA.
// S^T = K*Q^T so the softmax row (query) lives in the lane dimension.
__global__ void __launch_bounds__(256) k_attn(const u16* __restrict__ Qb,
                                              const u16* __restrict__ Kb,
                                              const u16* __restrict__ Vt,
                                              u16* __restrict__ Ob) {
  __shared__ __align__(16) u16 Ks[2][32 * HD];  // [key][d]      8KB x2
  __shared__ __align__(16) u16 Vs[2][HD * 32];  // [d][key]      8KB x2
  __shared__ __align__(16) u16 Pl[8][16 * 32];  // per-wave P^T patch

  const int tid = threadIdx.x;
  const int wave = tid >> 5, lane = tid & 31;
  const int half = lane >> 4, l16 = lane & 15;
  const int b  = blockIdx.x >> 8;
  const int h  = (blockIdx.x >> 4) & (NH - 1);
  const int qg = blockIdx.x & 15;
  const int kvh = h >> 2;  // GQA: 4 query heads per kv head
  const int q0 = qg * 128 + wave * 16;

  const u16* kb_base = Kb + (size_t)(b * NKV + kvh) * S * HD;
  const u16* vt_base = Vt + (size_t)(b * NKV + kvh) * HD * S;

  // Q^T as B-operand: lane col = query row q0+l16, contiguous d per lane
  const u16* qrow = Qb + ((size_t)(b * NH + h) * S + q0 + l16) * HD;
  v16bf qf[4];
#pragma unroll
  for (int c = 0; c < 4; ++c) qf[c] = load_b_frag(qrow, c * 32, half);

  v8f acc[8];
#pragma unroll
  for (int t = 0; t < 8; ++t) acc[t] = v8f{};
  float m = -1e30f, l = 0.f;
  const float scale = 0.08838834764831845f;  // 1/sqrt(128)
  const float LOG2E = 1.4426950408889634f;
  u16* pl = &Pl[wave][0];

  // cooperative async stage of one 32-key K/V tile (4 async ops per thread)
  auto stage = [&](int kb, int buf) {
#pragma unroll
    for (int j = 0; j < 2; ++j) {
      const int c = tid + j * 256;  // 512 x 16B chunks each
      async_copy16(kb_base + (size_t)(kb + (c >> 4)) * HD + (c & 15) * 8,
                   &Ks[buf][c * 8]);
      async_copy16(vt_base + (size_t)(c >> 2) * S + kb + (c & 3) * 8,
                   &Vs[buf][c * 8]);
    }
  };

  stage(0, 0);
  for (int kb = 0; kb < S; kb += 32) {
    const int cur = (kb >> 5) & 1;
    const int nkb = (kb + 32 < S) ? kb + 32 : 0;  // always issue 4 -> wait<=4 is exact
    stage(nkb, cur ^ 1);
    async_wait_le4();   // previous tile complete; next tile still in flight
    __syncthreads();

    // ---- S^T tiles from LDS K: D[M=key][N=query]
    v8f st0 = v8f{};
    v8f st1 = v8f{};
    const u16* kr0 = &Ks[cur][(size_t)l16 * HD];
    const u16* kr1 = &Ks[cur][(size_t)(16 + l16) * HD];
#pragma unroll
    for (int c = 0; c < 4; ++c) st0 = wmma_bf16(load_a_frag(kr0, c * 32, half), qf[c], st0);
#pragma unroll
    for (int c = 0; c < 4; ++c) st1 = wmma_bf16(load_a_frag(kr1, c * 32, half), qf[c], st1);

    // ---- online softmax (per-query stats in lane%16; combine halves via SWAPX16)
    float bm = -1e30f;
#pragma unroll
    for (int r = 0; r < 8; ++r) {
      st0[r] *= scale;
      st1[r] *= scale;
      bm = fmaxf(bm, fmaxf(st0[r], st1[r]));
    }
    bm = fmaxf(bm, xor16f(bm));
    const float mnew  = fmaxf(m, bm);
    const float alpha = __builtin_exp2f((m - mnew) * LOG2E);
    float bs = 0.f;
#pragma unroll
    for (int r = 0; r < 8; ++r) {
      const float p0 = __builtin_exp2f((st0[r] - mnew) * LOG2E);
      const float p1 = __builtin_exp2f((st1[r] - mnew) * LOG2E);
      st0[r] = p0;
      st1[r] = p1;
      bs += p0 + p1;
    }
    bs += xor16f(bs);
    l = l * alpha + bs;
    m = mnew;
#pragma unroll
    for (int t = 0; t < 8; ++t) acc[t] = acc[t] * alpha;

    // ---- P^T D-frag -> per-wave LDS patch [q][k] -> B-frag (32B contiguous/lane)
    uint4 w0, w1;
    w0.x = (unsigned)f2bf(st0[0]) | ((unsigned)f2bf(st0[1]) << 16);
    w0.y = (unsigned)f2bf(st0[2]) | ((unsigned)f2bf(st0[3]) << 16);
    w0.z = (unsigned)f2bf(st0[4]) | ((unsigned)f2bf(st0[5]) << 16);
    w0.w = (unsigned)f2bf(st0[6]) | ((unsigned)f2bf(st0[7]) << 16);
    w1.x = (unsigned)f2bf(st1[0]) | ((unsigned)f2bf(st1[1]) << 16);
    w1.y = (unsigned)f2bf(st1[2]) | ((unsigned)f2bf(st1[3]) << 16);
    w1.z = (unsigned)f2bf(st1[4]) | ((unsigned)f2bf(st1[5]) << 16);
    w1.w = (unsigned)f2bf(st1[6]) | ((unsigned)f2bf(st1[7]) << 16);
    *(uint4*)(pl + l16 * 32 + 8 * half)      = w0;  // k = 8*half + r      (tile0)
    *(uint4*)(pl + l16 * 32 + 16 + 8 * half) = w1;  // k = 16 + 8*half + r (tile1)

    FragAB pf;
    pf.q[0] = *(const uint4*)(pl + l16 * 32 + 16 * half);
    pf.q[1] = *(const uint4*)(pl + l16 * 32 + 16 * half + 8);

    // ---- O^T += V^T * P^T : 8 M-tiles cover d=0..127, V^T from LDS
#pragma unroll
    for (int t = 0; t < 8; ++t) {
      const u16* vr = &Vs[cur][(size_t)(t * 16 + l16) * 32];
      acc[t] = wmma_bf16(load_a_frag(vr, 0, half), pf.v, acc[t]);
    }
    __syncthreads();  // all waves done reading before next overwrite
  }

  // ---- normalize and store: lane owns query q0+l16; regs sweep d
  const float inv = 1.0f / l;
  u16* orow = Ob + (size_t)(b * S + q0 + l16) * HID + h * HD;
#pragma unroll
  for (int t = 0; t < 8; ++t)
#pragma unroll
    for (int r = 0; r < 8; ++r)
      orow[t * 16 + r + 8 * half] = f2bf(acc[t][r] * inv);
}

// ---------------------------------------------------------------- launcher
extern "C" void kernel_launch(void* const* d_in, const int* in_sizes, int n_in,
                              void* d_out, int out_size, void* d_ws, size_t ws_size,
                              hipStream_t stream) {
  const float* H  = (const float*)d_in[0];
  const float* Cs = (const float*)d_in[1];
  const float* Sn = (const float*)d_in[2];
  const float* Wq = (const float*)d_in[3];
  const float* Wk = (const float*)d_in[4];
  const float* Wv = (const float*)d_in[5];
  const float* Wo = (const float*)d_in[6];
  float* out = (float*)d_out;
  (void)in_sizes; (void)n_in; (void)out_size; (void)ws_size;

  char* ws = (char*)d_ws;
  size_t off = 0;
  auto take = [&](size_t bytes) -> char* {
    char* p = ws + off;
    off += (bytes + 255) & ~(size_t)255;
    return p;
  };
  u16* Hb   = (u16*)take((size_t)ROWS * HID * 2);   // hidden bf16
  u16* Wt   = (u16*)take((size_t)NQKV * HID * 2);   // [WqT|WkT|WvT] rows
  u16* WoT  = (u16*)take((size_t)HID * HID * 2);
  u16* QKVr = (u16*)take((size_t)ROWS * NQKV * 2);  // raw projections
  u16* Qb   = (u16*)take((size_t)BB * NH * S * HD * 2);
  u16* Kb   = (u16*)take((size_t)BB * NKV * S * HD * 2);
  u16* Vt   = (u16*)take((size_t)BB * NKV * HD * S * 2);
  u16* Ob   = (u16*)take((size_t)ROWS * HID * 2);

  k_f2bf<<<2048, 256, 0, stream>>>(H, Hb, (size_t)ROWS * HID);
  k_transpose<<<2048, 256, 0, stream>>>(Wq, Wt, HID, HID);
  k_transpose<<<512, 256, 0, stream>>>(Wk, Wt + (size_t)HID * HID, HID, NKV * HD);
  k_transpose<<<512, 256, 0, stream>>>(Wv, Wt + (size_t)(HID + NKV * HD) * HID, HID, NKV * HD);
  k_transpose<<<2048, 256, 0, stream>>>(Wo, WoT, HID, HID);

  k_gemm<NQKV, false><<<(ROWS / 16) * (NQKV / 512), 256, 0, stream>>>(Hb, Wt, QKVr);
  k_rope<<<BB * S, 256, 0, stream>>>(QKVr, Cs, Sn, Qb, Kb, Vt);
  k_attn<<<BB * NH * (S / 128), 256, 0, stream>>>(Qb, Kb, Vt, Ob);
  k_gemm<HID, true><<<(ROWS / 16) * (HID / 512), 256, 0, stream>>>(Ob, WoT, out);
}
